// FPN_LSS_Mamba_FO_14370960572907
// MI455X (gfx1250) — compile-verified, matrix operand order checked by reading, not run on hardware
//
#include <hip/hip_runtime.h>

// ---------------------------------------------------------------------------
// CDNA5 (gfx1250) wave32 + WMMA bf16 implementation of the FPN-LSS-Mamba head.
// All dense ops (projections, implicit-GEMM 3x3 convs, final 1x1) run through
// v_wmma_f32_16x16x32_bf16 with native v_cvt_pk_bf16_f32 conversion. Each wave
// owns a 32x32 output tile (4 WMMAs per K-step, 2 A + 2 B fragments).
// Selective scan is a VALU kernel, 1 lane per channel, SSM state (N=16) in
// VGPRs, dt-projection fused, cross-scan/merge via index permutation +
// f32 global atomics.
// ---------------------------------------------------------------------------

typedef __bf16 bf16_t;
typedef bf16_t v16bf __attribute__((ext_vector_type(16)));
typedef float  v8f   __attribute__((ext_vector_type(8)));

union FragBF { v16bf v; bf16_t e[16]; };

__device__ inline void ld8bf(bf16_t* d, const float* __restrict__ s) {
  const float4* p4 = reinterpret_cast<const float4*>(s);
  float4 a = p4[0], b = p4[1];
  d[0] = (bf16_t)a.x; d[1] = (bf16_t)a.y; d[2] = (bf16_t)a.z; d[3] = (bf16_t)a.w;
  d[4] = (bf16_t)b.x; d[5] = (bf16_t)b.y; d[6] = (bf16_t)b.z; d[7] = (bf16_t)b.w;
}

__device__ inline void zero8bf(bf16_t* d) {
  #pragma unroll
  for (int i = 0; i < 8; i++) d[i] = (bf16_t)0.f;
}

__device__ inline v8f wmma_bf16(const FragBF& a, const FragBF& b, v8f c) {
  return __builtin_amdgcn_wmma_f32_16x16x32_bf16(
      false, a.v, false, b.v, (short)0, c, false, false);
}

// cross-scan row map: scan-order index m of direction k -> row in row-major (H,W)
__device__ inline int scanRowMap(int m, int k, int H, int W) {
  int L = H * W;
  if (k == 0) return m;
  if (k == 2) return L - 1 - m;
  int lp = (k == 1) ? m : (L - 1 - m);   // col-major index l' = w*H + h
  int h = lp % H, w = lp / H;
  return h * W + w;
}

// ---------------------------------------------------------------------------
// Generic WMMA GEMM:  out[m,n] = sum_k A[row(m),k] * Wt[n,k]  (+ epilogue)
// Block = 128 threads (4 waves in a 2x2 grid); block tile 64(M) x 64(N);
// each wave: 32x32 via 4 WMMAs per 32-wide K step.
// permK==4 enables cross-scan A-row permutation (batch z = b*4 + k).
// epi: 0=none, 1=+bias, 2=+bias+residual, 3=BN(g,b)+ReLU
// ---------------------------------------------------------------------------
struct GemmP {
  const float* A;  long long lda, aBS;
  const float* Wt; long long wBS;
  const float* bias; const float* bng; const float* bnb;
  const float* resid;
  float* out; long long ldc, cBS;
  int M, N, K, H, W, permK, epi;
};

__global__ __launch_bounds__(128) void gemm_wmma(GemmP p) {
  int z = blockIdx.z;
  long long aOff, wOff, cOff; int kdir = 0;
  if (p.permK) {
    int b = z / p.permK; kdir = z % p.permK;
    aOff = (long long)b * p.aBS;
    wOff = (long long)kdir * p.wBS;
    cOff = (long long)z * p.cBS;
  } else {
    aOff = (long long)z * p.aBS; wOff = 0; cOff = (long long)z * p.cBS;
  }
  const float* A  = p.A + aOff;
  const float* Wm = p.Wt + wOff;
  float* Out = p.out + cOff;

  int wave = threadIdx.x >> 5, lane = threadIdx.x & 31;
  int wvM = wave >> 1, wvN = wave & 1;
  int row16 = lane & 15, hi = lane >> 4;
  int mBase = blockIdx.y * 64 + wvM * 32;
  int nBase = blockIdx.x * 64 + wvN * 32;

  int m0 = mBase + row16, m1 = mBase + 16 + row16;
  long long ar0 = p.permK ? (long long)scanRowMap(m0, kdir, p.H, p.W) : (long long)m0;
  long long ar1 = p.permK ? (long long)scanRowMap(m1, kdir, p.H, p.W) : (long long)m1;
  const float* Arow0 = A + ar0 * p.lda;
  const float* Arow1 = A + ar1 * p.lda;
  int n0 = nBase + row16, n1 = nBase + 16 + row16;
  const float* Wr0 = Wm + (long long)n0 * p.K;
  const float* Wr1 = Wm + (long long)n1 * p.K;
  bool v0 = n0 < p.N, v1 = n1 < p.N;

  v8f acc00 = (v8f){0.f,0.f,0.f,0.f,0.f,0.f,0.f,0.f};
  v8f acc01 = acc00, acc10 = acc00, acc11 = acc00;

  for (int k0 = 0; k0 < p.K; k0 += 32) {
    int ka = k0 + hi * 8;
    if (k0 + 32 < p.K) {
      // WGP-scope prefetch of the next K-chunk of the streamed A operand
      __builtin_prefetch(Arow0 + ka + 32, 0, 3);
      __builtin_prefetch(Arow1 + ka + 32, 0, 3);
    }
    FragBF a0, a1, b0, b1;
    ld8bf(a0.e,     Arow0 + ka);
    ld8bf(a0.e + 8, Arow0 + ka + 16);
    ld8bf(a1.e,     Arow1 + ka);
    ld8bf(a1.e + 8, Arow1 + ka + 16);
    if (v0) { ld8bf(b0.e, Wr0 + ka); ld8bf(b0.e + 8, Wr0 + ka + 16); }
    else    { zero8bf(b0.e); zero8bf(b0.e + 8); }
    if (v1) { ld8bf(b1.e, Wr1 + ka); ld8bf(b1.e + 8, Wr1 + ka + 16); }
    else    { zero8bf(b1.e); zero8bf(b1.e + 8); }
    acc00 = wmma_bf16(a0, b0, acc00);
    acc01 = wmma_bf16(a0, b1, acc01);
    acc10 = wmma_bf16(a1, b0, acc10);
    acc11 = wmma_bf16(a1, b1, acc11);
  }

  v8f accs[2][2] = {{acc00, acc01}, {acc10, acc11}};
  #pragma unroll
  for (int mi = 0; mi < 2; mi++) {
    #pragma unroll
    for (int sub = 0; sub < 2; sub++) {
      int n = nBase + sub * 16 + row16;
      if (n >= p.N) continue;
      #pragma unroll
      for (int e = 0; e < 8; e++) {
        int mo = mBase + mi * 16 + hi * 8 + e;
        float v = accs[mi][sub][e];
        if (p.epi == 1)      v += p.bias[n];
        else if (p.epi == 2) v += p.bias[n] + p.resid[(long long)mo * p.ldc + n];
        else if (p.epi == 3) v = fmaxf(0.f, v * (p.bng[n] * rsqrtf(1.f + 1e-5f)) + p.bnb[n]);
        Out[(long long)mo * p.ldc + n] = v;
      }
    }
  }
}

// ---------------------------------------------------------------------------
// 3x3 pad=1 conv as implicit GEMM over NHWC; K = 9*Cin, kk = tap*Cin + ci.
// Same 2x2-wave 64x64 block tile. Epilogue: inference BN + ReLU.
// ---------------------------------------------------------------------------
struct ConvP {
  const float* in; const float* Wt; const float* g; const float* b;
  float* out;
  int B, H, W, Cin, Cout;
};

__device__ inline void ldA_conv8(bf16_t* d, const float* __restrict__ in,
                                 int b, int h, int w, int H, int W, int Cin, int kk) {
  int tap = kk / Cin, ci = kk - tap * Cin;     // 8-run stays inside one tap (Cin%8==0)
  int dh = tap / 3 - 1, dw = tap - (tap / 3) * 3 - 1;
  int hh = h + dh, ww = w + dw;
  if ((unsigned)hh < (unsigned)H && (unsigned)ww < (unsigned)W) {
    ld8bf(d, in + (((long long)b * H + hh) * W + ww) * Cin + ci);
  } else {
    zero8bf(d);
  }
}

__device__ inline void ldW_conv8(bf16_t* d, const float* __restrict__ Wt,
                                 int n, int Cin, int kk) {
  int tap = kk / Cin, ci = kk - tap * Cin;
  const float* s = Wt + ((long long)n * Cin + ci) * 9 + tap;
  #pragma unroll
  for (int i = 0; i < 8; i++) d[i] = (bf16_t)s[(long long)i * 9];
}

__global__ __launch_bounds__(128) void conv3x3_wmma(ConvP p) {
  int wave = threadIdx.x >> 5, lane = threadIdx.x & 31;
  int wvM = wave >> 1, wvN = wave & 1;
  int row16 = lane & 15, hi = lane >> 4;
  int mBase = blockIdx.y * 64 + wvM * 32;
  int nBase = blockIdx.x * 64 + wvN * 32;

  int m0 = mBase + row16, m1 = mBase + 16 + row16;
  int w0 = m0 % p.W; int t0 = m0 / p.W; int h0 = t0 % p.H; int b0i = t0 / p.H;
  int w1 = m1 % p.W; int t1 = m1 / p.W; int h1 = t1 % p.H; int b1i = t1 / p.H;
  int K = 9 * p.Cin;
  int n0 = nBase + row16, n1 = nBase + 16 + row16;

  v8f acc00 = (v8f){0.f,0.f,0.f,0.f,0.f,0.f,0.f,0.f};
  v8f acc01 = acc00, acc10 = acc00, acc11 = acc00;

  for (int k0 = 0; k0 < K; k0 += 32) {
    int ka = k0 + hi * 8;
    FragBF a0, a1, b0, b1;
    ldA_conv8(a0.e,     p.in, b0i, h0, w0, p.H, p.W, p.Cin, ka);
    ldA_conv8(a0.e + 8, p.in, b0i, h0, w0, p.H, p.W, p.Cin, ka + 16);
    ldA_conv8(a1.e,     p.in, b1i, h1, w1, p.H, p.W, p.Cin, ka);
    ldA_conv8(a1.e + 8, p.in, b1i, h1, w1, p.H, p.W, p.Cin, ka + 16);
    ldW_conv8(b0.e,     p.Wt, n0, p.Cin, ka);
    ldW_conv8(b0.e + 8, p.Wt, n0, p.Cin, ka + 16);
    ldW_conv8(b1.e,     p.Wt, n1, p.Cin, ka);
    ldW_conv8(b1.e + 8, p.Wt, n1, p.Cin, ka + 16);
    acc00 = wmma_bf16(a0, b0, acc00);
    acc01 = wmma_bf16(a0, b1, acc01);
    acc10 = wmma_bf16(a1, b0, acc10);
    acc11 = wmma_bf16(a1, b1, acc11);
  }

  v8f accs[2][2] = {{acc00, acc01}, {acc10, acc11}};
  #pragma unroll
  for (int mi = 0; mi < 2; mi++) {
    #pragma unroll
    for (int sub = 0; sub < 2; sub++) {
      int n = nBase + sub * 16 + row16;
      float sc = p.g[n] * rsqrtf(1.f + 1e-5f);
      float sh = p.b[n];
      #pragma unroll
      for (int e = 0; e < 8; e++) {
        int mo = mBase + mi * 16 + hi * 8 + e;
        float v = fmaxf(0.f, accs[mi][sub][e] * sc + sh);
        p.out[(long long)mo * p.Cout + n] = v;
      }
    }
  }
}

// ---------------------------------------------------------------------------
// LayerNorm over channel dim (NHWC rows); optional SiLU(z) gating.
// ---------------------------------------------------------------------------
__global__ __launch_bounds__(256) void ln_kernel(
    const float* __restrict__ in, long long ldin,
    const float* __restrict__ g, const float* __restrict__ bta,
    const float* __restrict__ gate, long long ldg,
    float* __restrict__ out, long long ldout, int C) {
  long long row = blockIdx.x;
  const float* x = in + row * ldin;
  float s = 0.f, s2 = 0.f;
  for (int c = threadIdx.x; c < C; c += 256) { float v = x[c]; s += v; s2 += v * v; }
  __shared__ float rs[8], rs2[8];
  for (int o = 16; o > 0; o >>= 1) { s += __shfl_xor(s, o, 32); s2 += __shfl_xor(s2, o, 32); }
  int wv = threadIdx.x >> 5;
  if ((threadIdx.x & 31) == 0) { rs[wv] = s; rs2[wv] = s2; }
  __syncthreads();
  if (threadIdx.x == 0) {
    float a = 0.f, b = 0.f;
    for (int i = 0; i < 8; i++) { a += rs[i]; b += rs2[i]; }
    rs[0] = a; rs2[0] = b;
  }
  __syncthreads();
  float mu  = rs[0] / C;
  float var = rs2[0] / C - mu * mu;
  float inv = rsqrtf(var + 1e-5f);
  const float* zg = gate ? gate + row * ldg : nullptr;
  float* o = out + row * ldout;
  for (int c = threadIdx.x; c < C; c += 256) {
    float v = (x[c] - mu) * inv * g[c] + bta[c];
    if (zg) { float z = zg[c]; v *= z / (1.f + __expf(-z)); }
    o[c] = v;
  }
}

// ---------------------------------------------------------------------------
// Depthwise 3x3 conv + bias + SiLU; input = first C columns of xz (stride 2C).
// ---------------------------------------------------------------------------
__global__ void dwconv_silu(const float* __restrict__ xz, const float* __restrict__ wt,
                            const float* __restrict__ bias, float* __restrict__ out,
                            int B, int H, int W, int C) {
  long long idx = (long long)blockIdx.x * blockDim.x + threadIdx.x;
  long long tot = (long long)B * H * W * C;
  if (idx >= tot) return;
  int c = idx % C; long long t = idx / C;
  int w = t % W; t /= W; int h = t % H; int b = t / H;
  float acc = bias[c];
  #pragma unroll
  for (int dh = -1; dh <= 1; dh++)
    #pragma unroll
    for (int dw = -1; dw <= 1; dw++) {
      int hh = h + dh, ww = w + dw;
      if ((unsigned)hh < (unsigned)H && (unsigned)ww < (unsigned)W) {
        float xi = xz[(((long long)b * H + hh) * W + ww) * (2LL * C) + c];
        acc += xi * wt[c * 9 + (dh + 1) * 3 + (dw + 1)];
      }
    }
  out[idx] = acc / (1.f + __expf(-acc));
}

// ---------------------------------------------------------------------------
// Selective scan: 1 lane per channel, 16-state in VGPRs, dt-proj fused.
// grid = (C/128, 4, B), block = 128 (4 waves). Accumulates into ym via atomics.
// ---------------------------------------------------------------------------
__global__ __launch_bounds__(128) void scan_kernel(
    const float* __restrict__ xc, const float* __restrict__ xdbl,
    const float* __restrict__ dt_w, const float* __restrict__ dt_b,
    const float* __restrict__ A_log, const float* __restrict__ Dp,
    float* __restrict__ ym, int B, int H, int W, int C, int R) {
  int k = blockIdx.y, b = blockIdx.z;
  int L = H * W, RW = R + 32;
  int d = blockIdx.x * 128 + threadIdx.x;
  long long kd = (long long)k * C + d;

  float dtw[40];
  for (int r = 0; r < R; r++) dtw[r] = dt_w[kd * R + r];
  float dtb = dt_b[kd];
  float An[16];
  #pragma unroll
  for (int n = 0; n < 16; n++) An[n] = -__expf(A_log[kd * 16 + n]);
  float Dd = Dp[kd];
  float h[16];
  #pragma unroll
  for (int n = 0; n < 16; n++) h[n] = 0.f;

  __shared__ float sh[72];
  const float* xrow  = xc + (long long)b * L * C;
  const float* xdrow = xdbl + (long long)(b * 4 + k) * L * RW;
  float* ymb = ym + (long long)b * L * C;

  for (int l = 0; l < L; l++) {
    if (threadIdx.x < RW) sh[threadIdx.x] = xdrow[(long long)l * RW + threadIdx.x];
    __syncthreads();
    float dtr = dtb;
    for (int r = 0; r < R; r++) dtr += sh[r] * dtw[r];
    float dt = (dtr > 20.f) ? dtr : log1pf(__expf(dtr));
    int rm = scanRowMap(l, k, H, W);
    float xv = xrow[(long long)rm * C + d];
    float dx = dt * xv;
    float y = 0.f;
    #pragma unroll
    for (int n = 0; n < 16; n++) {
      float dA = __expf(dt * An[n]);
      float hn = h[n] * dA + dx * sh[R + n];
      h[n] = hn;
      y += hn * sh[R + 16 + n];
    }
    y += Dd * xv;
    atomicAdd(&ymb[(long long)rm * C + d], y);
    __syncthreads();
  }
}

// ---------------------------------------------------------------------------
// Elementwise helpers: transposes, bilinear upsample (align_corners), concat.
// ---------------------------------------------------------------------------
__global__ void nchw_to_nhwc(const float* __restrict__ in, float* __restrict__ out,
                             int B, int C, int H, int W) {
  long long idx = (long long)blockIdx.x * blockDim.x + threadIdx.x;
  long long tot = (long long)B * H * W * C;
  if (idx >= tot) return;
  int c = idx % C; long long t = idx / C;
  int w = t % W; t /= W; int h = t % H; int b = t / H;
  out[idx] = in[(((long long)b * C + c) * H + h) * W + w];
}

__global__ void nhwc_to_nchw(const float* __restrict__ in, float* __restrict__ out,
                             int B, int C, int H, int W) {
  long long idx = (long long)blockIdx.x * blockDim.x + threadIdx.x;
  long long tot = (long long)B * C * H * W;
  if (idx >= tot) return;
  int w = idx % W; long long t = idx / W;
  int h = t % H; t /= H; int c = t % C; int b = t / C;
  out[idx] = in[(((long long)b * H + h) * W + w) * C + c];
}

__global__ void upsample_nhwc(const float* __restrict__ in, float* __restrict__ out,
                              int B, int Hin, int Win, int C, int Hout, int Wout) {
  long long idx = (long long)blockIdx.x * blockDim.x + threadIdx.x;
  long long tot = (long long)B * Hout * Wout * C;
  if (idx >= tot) return;
  int c = idx % C; long long t = idx / C;
  int wo = t % Wout; t /= Wout; int ho = t % Hout; int b = t / Hout;
  float ph = (float)ho * (float)(Hin - 1) / (float)(Hout - 1);
  float pw = (float)wo * (float)(Win - 1) / (float)(Wout - 1);
  int i0 = (int)ph; if (i0 > Hin - 2) i0 = Hin - 2; float fh = ph - i0;
  int j0 = (int)pw; if (j0 > Win - 2) j0 = Win - 2; float fw = pw - j0;
  const float* base = in + (((long long)b * Hin + i0) * Win + j0) * C + c;
  float v00 = base[0], v01 = base[C];
  float v10 = base[(long long)Win * C], v11 = base[(long long)Win * C + C];
  out[idx] = (1.f - fh) * ((1.f - fw) * v00 + fw * v01) +
             fh * ((1.f - fw) * v10 + fw * v11);
}

__global__ void concat_kernel(const float* __restrict__ f0 /*NCHW B,128,64,64*/,
                              const float* __restrict__ up /*NHWC B,64,64,512*/,
                              float* __restrict__ out /*NHWC B,64,64,640*/, int B) {
  long long idx = (long long)blockIdx.x * blockDim.x + threadIdx.x;
  long long tot = (long long)B * 64 * 64 * 640;
  if (idx >= tot) return;
  int c = idx % 640; long long t = idx / 640;
  int w = t % 64; t /= 64; int h = t % 64; int b = t / 64;
  float v;
  if (c < 128) v = f0[(((long long)b * 128 + c) * 64 + h) * 64 + w];
  else         v = up[(((long long)b * 64 + h) * 64 + w) * 512 + (c - 128)];
  out[idx] = v;
}

// ---------------------------------------------------------------------------
// Host-side orchestration
// ---------------------------------------------------------------------------
struct WsBuf {
  float *aX, *xz, *ln, *xc, *xdbl, *ym, *gated;
  float *x1up, *convA, *convB, *up2, *c3out, *finN;
};

// block param order within a block dict after alphabetical pytree flatten:
// 0 A_log, 1 D, 2 dt_b, 3 dt_w, 4 dw_b, 5 dw_w, 6 in_b, 7 in_w,
// 8 ln_b, 9 ln_g, 10 on_b, 11 on_g, 12 out_b, 13 out_w, 14 xp_w
static void runVSS(hipStream_t s, const float* const* P, int base, const WsBuf& w,
                   int B, int H, int Wd, int C, int R) {
  int L = H * Wd, M = B * L, RW = R + 32;

  ln_kernel<<<M, 256, 0, s>>>(w.aX, C, P[base + 9], P[base + 8], nullptr, 0, w.ln, C, C);

  GemmP g1{};
  g1.A = w.ln; g1.lda = C; g1.aBS = 0; g1.Wt = P[base + 7]; g1.wBS = 0;
  g1.bias = P[base + 6]; g1.out = w.xz; g1.ldc = 2 * C; g1.cBS = 0;
  g1.M = M; g1.N = 2 * C; g1.K = C; g1.H = H; g1.W = Wd; g1.permK = 0; g1.epi = 1;
  gemm_wmma<<<dim3((2 * C + 63) / 64, M / 64, 1), 128, 0, s>>>(g1);

  long long tot = (long long)M * C;
  dwconv_silu<<<(unsigned)((tot + 255) / 256), 256, 0, s>>>(
      w.xz, P[base + 5], P[base + 4], w.xc, B, H, Wd, C);

  hipMemsetAsync(w.ym, 0, (size_t)tot * sizeof(float), s);

  GemmP g2{};
  g2.A = w.xc; g2.lda = C; g2.aBS = (long long)L * C;
  g2.Wt = P[base + 14]; g2.wBS = (long long)RW * C;
  g2.out = w.xdbl; g2.ldc = RW; g2.cBS = (long long)L * RW;
  g2.M = L; g2.N = RW; g2.K = C; g2.H = H; g2.W = Wd; g2.permK = 4; g2.epi = 0;
  gemm_wmma<<<dim3((RW + 63) / 64, L / 64, B * 4), 128, 0, s>>>(g2);

  scan_kernel<<<dim3(C / 128, 4, B), 128, 0, s>>>(
      w.xc, w.xdbl, P[base + 3], P[base + 2], P[base + 0], P[base + 1],
      w.ym, B, H, Wd, C, R);

  ln_kernel<<<M, 256, 0, s>>>(w.ym, C, P[base + 11], P[base + 10],
                              w.xz + C, 2 * C, w.gated, C, C);

  GemmP g3{};
  g3.A = w.gated; g3.lda = C; g3.Wt = P[base + 13]; g3.bias = P[base + 12];
  g3.resid = w.aX; g3.out = w.aX; g3.ldc = C;
  g3.M = M; g3.N = C; g3.K = C; g3.H = H; g3.W = Wd; g3.permK = 0; g3.epi = 2;
  gemm_wmma<<<dim3((C + 63) / 64, M / 64, 1), 128, 0, s>>>(g3);
}

extern "C" void kernel_launch(void* const* d_in, const int* in_sizes, int n_in,
                              void* d_out, int out_size, void* d_ws, size_t ws_size,
                              hipStream_t stream) {
  if (n_in < 103) return;
  const float* P[103];
  for (int i = 0; i < 103; i++) P[i] = (const float*)d_in[i];
  // top-level alphabetical flatten:
  // 0 feat0, 1 feat2, 2 bn1_b, 3 bn1_g, 4 bn2_b, 5 bn2_g, 6 bn3_b, 7 bn3_g,
  // 8 ccw1[15], 23 ccw2[15], 38 ccw3[15], 53 conv1_w, 54 conv2_w,
  // 55 cw1[15], 70 cw2[15], 85 cw3[15], 100 final_b, 101 final_w, 102 up_conv_w

  float* W = (float*)d_ws;
  size_t off = 0;
  auto alloc = [&](size_t n) { size_t o = off; off += (n + 1023) & ~(size_t)1023; return o; };
  size_t oAX   = alloc(5242880);   // x NHWC, max (2,64,64,640)
  size_t oXZ   = alloc(10485760);  // (M, 2C)
  size_t oLN   = alloc(5242880);
  size_t oXC   = alloc(5242880);
  size_t oXD   = alloc(2359296);   // (B,4,L,72)
  size_t oYM   = alloc(5242880);
  size_t oGT   = alloc(5242880);
  size_t oUP1  = alloc(4194304);   // (2,64,64,512)
  size_t oCA   = alloc(4194304);
  size_t oCB   = alloc(4194304);
  size_t oUP2  = alloc(16777216);  // (2,128,128,512)
  size_t oC3   = alloc(8388608);   // (2,128,128,256)
  size_t oFN   = alloc(8388608);
  if (ws_size < off * sizeof(float)) return;

  WsBuf w;
  w.aX = W + oAX; w.xz = W + oXZ; w.ln = W + oLN; w.xc = W + oXC;
  w.xdbl = W + oXD; w.ym = W + oYM; w.gated = W + oGT;
  w.x1up = W + oUP1; w.convA = W + oCA; w.convB = W + oCB;
  w.up2 = W + oUP2; w.c3out = W + oC3; w.finN = W + oFN;

  const int B = 2;

  // deep path: feat2 NCHW (2,512,16,16) -> NHWC
  {
    long long tot = (long long)B * 512 * 16 * 16;
    nchw_to_nhwc<<<(unsigned)((tot + 255) / 256), 256, 0, stream>>>(P[1], w.aX, B, 512, 16, 16);
  }
  runVSS(stream, P, 8,  w, B, 16, 16, 512, 32);   // ccw1
  runVSS(stream, P, 55, w, B, 16, 16, 512, 32);   // cw1
  runVSS(stream, P, 23, w, B, 16, 16, 512, 32);   // ccw2
  runVSS(stream, P, 70, w, B, 16, 16, 512, 32);   // cw2

  // upsample x4 (16->64) + concat with feat0 -> (B,64,64,640) NHWC
  {
    long long tot = (long long)B * 64 * 64 * 512;
    upsample_nhwc<<<(unsigned)((tot + 255) / 256), 256, 0, stream>>>(
        w.aX, w.x1up, B, 16, 16, 512, 64, 64);
    long long tc = (long long)B * 64 * 64 * 640;
    concat_kernel<<<(unsigned)((tc + 255) / 256), 256, 0, stream>>>(P[0], w.x1up, w.aX, B);
  }
  runVSS(stream, P, 38, w, B, 64, 64, 640, 40);   // ccw3
  runVSS(stream, P, 85, w, B, 64, 64, 640, 40);   // cw3

  // conv1 640->512 @64x64, BN+ReLU
  {
    ConvP c{ w.aX, P[53], P[3], P[2], w.convA, B, 64, 64, 640, 512 };
    conv3x3_wmma<<<dim3(512 / 64, (B * 64 * 64) / 64, 1), 128, 0, stream>>>(c);
  }
  // conv2 512->512 @64x64, BN+ReLU
  {
    ConvP c{ w.convA, P[54], P[5], P[4], w.convB, B, 64, 64, 512, 512 };
    conv3x3_wmma<<<dim3(512 / 64, (B * 64 * 64) / 64, 1), 128, 0, stream>>>(c);
  }
  // upsample x2 (64->128)
  {
    long long tot = (long long)B * 128 * 128 * 512;
    upsample_nhwc<<<(unsigned)((tot + 255) / 256), 256, 0, stream>>>(
        w.convB, w.up2, B, 64, 64, 512, 128, 128);
  }
  // up_conv 512->256 @128x128, BN+ReLU
  {
    ConvP c{ w.up2, P[102], P[7], P[6], w.c3out, B, 128, 128, 512, 256 };
    conv3x3_wmma<<<dim3(256 / 64, (B * 128 * 128) / 64, 1), 128, 0, stream>>>(c);
  }
  // final 1x1 conv 256->256 (+bias) as GEMM
  {
    GemmP g{};
    g.A = w.c3out; g.lda = 256; g.Wt = P[101]; g.bias = P[100];
    g.out = w.finN; g.ldc = 256;
    g.M = B * 128 * 128; g.N = 256; g.K = 256; g.H = 128; g.W = 128;
    g.permK = 0; g.epi = 1;
    gemm_wmma<<<dim3(256 / 64, (B * 128 * 128) / 64, 1), 128, 0, stream>>>(g);
  }
  // NHWC -> NCHW into d_out (2,256,128,128)
  {
    long long tot = (long long)B * 256 * 128 * 128;
    nhwc_to_nchw<<<(unsigned)((tot + 255) / 256), 256, 0, stream>>>(
        w.finN, (float*)d_out, B, 256, 128, 128);
  }
  (void)in_sizes; (void)out_size;
}
